// MODEL_8143257993816
// MI455X (gfx1250) — matrix operand order repaired and checked by reading, not compiled, not run on hardware
//
#include <hip/hip_runtime.h>
#include <hip/hip_fp16.h>

typedef __attribute__((ext_vector_type(16))) _Float16 v16h;
typedef __attribute__((ext_vector_type(8)))  float    v8f;
typedef int v4i __attribute__((vector_size(16)));

#define HID    64
#define DRUG_N 100000
#define DIS_N  100000
#define NRD_N  200000
#define LAYERS 3
#define EPSN   0.1f

#if __has_builtin(__builtin_amdgcn_global_load_async_to_lds_b128) && \
    __has_builtin(__builtin_amdgcn_s_wait_asynccnt)
#define GATE_ASYNC_LDS 1
#endif

union HFrag { v16h v; _Float16 h[16]; };
union FAcc  { v8f  v; float    f[8];  };

// ---------------------------------------------------------------- utilities
__global__ void zero_kernel(float* __restrict__ p, size_t n) {
  size_t i      = (size_t)blockIdx.x * blockDim.x + threadIdx.x;
  size_t stride = (size_t)gridDim.x * blockDim.x;
  for (; i < n; i += stride) p[i] = 0.f;
}

__global__ void degree_kernel(const int* __restrict__ row, const int* __restrict__ col,
                              const float* __restrict__ val,
                              float* __restrict__ deg_r, float* __restrict__ deg_c, int n) {
  int i = blockIdx.x * blockDim.x + threadIdx.x;
  if (i < n) {
    float v = val[i];
    atomicAdd(&deg_r[row[i]], v);
    atomicAdd(&deg_c[col[i]], v);
  }
}

__device__ __forceinline__ float wave_sum32(float v) {
#pragma unroll
  for (int m = 16; m >= 1; m >>= 1) v += __shfl_xor(v, m, 32);
  return v;
}

__device__ __forceinline__ float uhash(unsigned layer, unsigned idx) {
  unsigned h = idx * 0x9E3779B1u + layer * 0x85EBCA77u + 0x165667B1u;
  h ^= h >> 16; h *= 0x7FEB352Du;
  h ^= h >> 15; h *= 0x846CA68Bu;
  h ^= h >> 16;
  return (float)(h >> 8) * (1.0f / 16777216.0f);
}

__device__ __forceinline__ float sgnf(float x) {
  return (float)(x > 0.f) - (float)(x < 0.f);
}

// ---------------------------------------------- gated embedding: emb*sigmoid(emb@W+b)
// One wave computes a 16x64 output tile. K=64 -> two v_wmma_f32_16x16x32_f16
// per 16x16 N-tile, 4 N-tiles. W async-staged to LDS (f32), then swizzled into
// lane-contiguous f16 B-fragments so each fragment is two ds_load_b128.
__global__ void __launch_bounds__(256)
gate_wmma_kernel(const float* __restrict__ emb, const float* __restrict__ W,
                 const float* __restrict__ bias,
                 float* __restrict__ outA, float* __restrict__ outB, int nrows) {
  __shared__ float    sWraw[HID * HID];    // 16 KB raw W
  __shared__ _Float16 sWfrag[HID * HID];   // 8 KB fragment-ordered f16 W
  __shared__ float    sB[HID];

#ifdef GATE_ASYNC_LDS
  // Async copy 16 KB of W into LDS: 4 rounds x 256 lanes x 16 B (ASYNCcnt).
#pragma unroll
  for (int i = 0; i < 4; i++) {
    const int off = (i * 256 + (int)threadIdx.x) * 4;   // float index, 16B/lane
    __builtin_amdgcn_global_load_async_to_lds_b128(
        (v4i*)(W + off),
        (__attribute__((address_space(3))) v4i*)(sWraw + off), 0, 0);
  }
  __builtin_amdgcn_s_wait_asynccnt(0);
#else
  for (int i = threadIdx.x; i < HID * HID; i += 256) sWraw[i] = W[i];
#endif
  if (threadIdx.x < HID) sB[threadIdx.x] = bias[threadIdx.x];
  __syncthreads();

  // Swizzle+convert: fragment element idx=(blk*32+lane)*16+j holds W[K][N] with
// blk = nt*2+kblk, K = kblk*32 + (lane>=16)*8 + (j&7) + (j>>3)*16, N = nt*16 + lane%16
  // (ISA 16-bit B 32x16 layout).
#pragma unroll
  for (int t = 0; t < 16; t++) {
    const int idx = t * 256 + (int)threadIdx.x;
    const int j   = idx & 15;
    const int ln  = (idx >> 4) & 31;
    const int blk = idx >> 9;                         // 0..7
    const int K   = (blk & 1) * 32 + (ln >> 4) * 8 + (j & 7) + (j >> 3) * 16;
    const int N   = (blk >> 1) * 16 + (ln & 15);
    sWfrag[idx] = (_Float16)sWraw[K * HID + N];
  }
  __syncthreads();

  const int wave = threadIdx.x >> 5, lane = threadIdx.x & 31;
  const int tile = blockIdx.x * 8 + wave;
  if (tile >= (nrows >> 4)) return;   // wave-uniform exit; EXEC stays all-ones

  const int m  = lane & 15;   // row within tile (A) / column N (C/D)
  const int hi = lane >> 4;
  const int cb = hi * 8;      // A: K-run base per half-wave

  // A fragments (ISA 16-bit A 16x32 layout): four contiguous 8-float runs per
  // lane -> 8x global_load_b128, then pack to f16.
  const float4* ar = (const float4*)(emb + ((size_t)tile * 16 + m) * HID);
  const int     qb = cb >> 2;         // 0 or 2
  float4 q[8];
  q[0] = ar[qb];      q[1] = ar[qb + 1];     // K = cb..cb+7        (a0 lo)
  q[2] = ar[qb + 4];  q[3] = ar[qb + 5];     // K = 16+cb..16+cb+7  (a0 hi)
  q[4] = ar[qb + 8];  q[5] = ar[qb + 9];     // K = 32+cb..         (a1 lo)
  q[6] = ar[qb + 12]; q[7] = ar[qb + 13];    // K = 48+cb..         (a1 hi)
  const float* qf = (const float*)q;
  HFrag a0, a1;
#pragma unroll
  for (int j = 0; j < 8; j++) {
    a0.h[j]     = (_Float16)qf[j];
    a0.h[8 + j] = (_Float16)qf[8 + j];
    a1.h[j]     = (_Float16)qf[16 + j];
    a1.h[8 + j] = (_Float16)qf[24 + j];
  }

#pragma unroll
  for (int nt = 0; nt < 4; nt++) {
    const v16h b0 = *(const v16h*)(sWfrag + ((nt * 2 + 0) * 32 + lane) * 16);
    const v16h b1 = *(const v16h*)(sWfrag + ((nt * 2 + 1) * 32 + lane) * 16);
    v8f c = {};
    c = __builtin_amdgcn_wmma_f32_16x16x32_f16(false, a0.v, false, b0, (short)0, c, false, false);
    c = __builtin_amdgcn_wmma_f32_16x16x32_f16(false, a1.v, false, b1, (short)0, c, false, false);
    FAcc acc; acc.v = c;

    // C/D layout: VGPR r -> M = r + 8*hi ; N = lane&15
#pragma unroll
    for (int r = 0; r < 8; r++) {
      const int    M    = r + hi * 8;
      const int    N    = nt * 16 + m;
      const size_t rowi = (size_t)tile * 16 + M;
      const float  pre  = acc.f[r] + sB[N];
      const float  g    = 1.f / (1.f + __expf(-pre));
      const float  v    = emb[rowi * HID + N] * g;
      outA[rowi * HID + N] = v;
      outB[rowi * HID + N] = v;
    }
  }
}

// ---------------------------------------------------------------- SpMM (wave/edge)
__global__ void __launch_bounds__(256)
spmm_kernel(const float* __restrict__ x, const int* __restrict__ row,
            const int* __restrict__ col, const float* __restrict__ val,
            const float* __restrict__ dr, const float* __restrict__ dc,
            float* __restrict__ out, int nedges) {
  const int e = blockIdx.x * 8 + (threadIdx.x >> 5);
  if (e >= nedges) return;
  const int lane = threadIdx.x & 31;
  __builtin_prefetch(col + e + 2048, 0, 0);   // global_prefetch_b8 on edge stream
  const int   r  = row[e], c = col[e];
  const float vn = val[e] * __frsqrt_rn(fmaxf(dr[r] * dc[c], 1e-12f));
  const float2 xv = *(const float2*)(x + (size_t)c * HID + lane * 2);
  float* o = out + (size_t)r * HID + lane * 2;
  atomicAdd(o,     vn * xv.x);
  atomicAdd(o + 1, vn * xv.y);
}

// --------------------------------------------------------- rd init + stack slot 0
__global__ void init_rd_kernel(const float* __restrict__ de, const float* __restrict__ se,
                               float* __restrict__ cur_rd, float* __restrict__ rd_stack) {
  const size_t i = (size_t)blockIdx.x * blockDim.x + threadIdx.x;
  if (i >= (size_t)NRD_N * HID) return;
  const size_t n = i >> 6;
  const int    f = (int)(i & 63);
  const float  v = (n < DRUG_N) ? de[i] : se[i - (size_t)DRUG_N * HID];
  cur_rd[i] = v;
  rd_stack[n * (size_t)((LAYERS + 1) * HID) + f] = v;
}

// ---------------------------------- per-layer epilogue: noise, combine, l2n, stack
__global__ void __launch_bounds__(256)
layer_post_kernel(int layer, const float* __restrict__ d0, const float* __restrict__ s0,
                  const float* __restrict__ r0, float* __restrict__ cur_drug,
                  float* __restrict__ cur_dis, float* __restrict__ cur_rd,
                  float* __restrict__ sum_drug, float* __restrict__ sum_dis,
                  float* __restrict__ rd_stack) {
  const int n = blockIdx.x * 8 + (threadIdx.x >> 5);
  if (n >= NRD_N) return;
  const int    lane = threadIdx.x & 31;
  const int    f    = lane * 2;
  const size_t base = (size_t)n * HID + f;

  float rx = r0[base], ry = r0[base + 1];
  const float u0 = uhash((unsigned)layer, (unsigned)(n * HID + f));
  const float u1 = uhash((unsigned)layer, (unsigned)(n * HID + f + 1));
  const float ns = 1.f / fmaxf(__fsqrt_rn(wave_sum32(u0 * u0 + u1 * u1)), 1e-12f);
  rx += sgnf(rx) * u0 * ns * EPSN;
  ry += sgnf(ry) * u1 * ns * EPSN;

  const float rs = 1.f / fmaxf(__fsqrt_rn(wave_sum32(rx * rx + ry * ry)), 1e-12f);
  float* st = rd_stack + (size_t)n * ((LAYERS + 1) * HID) + (size_t)(layer + 1) * HID + f;
  st[0] = rx * rs;
  st[1] = ry * rs;

  if (n < DRUG_N) {
    const float dx = d0[base], dy = d0[base + 1];
    const float dsc = 1.f / fmaxf(__fsqrt_rn(wave_sum32(dx * dx + dy * dy)), 1e-12f);
    sum_drug[base]     += dx * dsc;
    sum_drug[base + 1] += dy * dsc;
    const float cx = 0.5f * dx + 0.5f * rx, cy = 0.5f * dy + 0.5f * ry;
    cur_drug[base] = cx; cur_drug[base + 1] = cy;
    cur_rd[base]   = cx; cur_rd[base + 1]   = cy;
  } else {
    const size_t b2 = base - (size_t)DRUG_N * HID;
    const float sx = s0[b2], sy = s0[b2 + 1];
    const float ssc = 1.f / fmaxf(__fsqrt_rn(wave_sum32(sx * sx + sy * sy)), 1e-12f);
    sum_dis[b2]     += sx * ssc;
    sum_dis[b2 + 1] += sy * ssc;
    const float cx = 0.5f * sx + 0.5f * rx, cy = 0.5f * sy + 0.5f * ry;
    cur_dis[b2]  = cx; cur_dis[b2 + 1]  = cy;
    cur_rd[base] = cx; cur_rd[base + 1] = cy;
  }
}

// ---------------------------------------------------------------- final outputs
__global__ void final_kernel(const float* __restrict__ de, const float* __restrict__ se,
                             const float* __restrict__ sum_drug,
                             const float* __restrict__ sum_dis, float* __restrict__ out) {
  const size_t i = (size_t)blockIdx.x * blockDim.x + threadIdx.x;
  const size_t S = (size_t)DRUG_N * HID;
  if (i >= S) return;
  const float dE = sum_drug[i] * 0.25f;   // mean over L+1 = 4 entries
  const float sE = sum_dis[i] * 0.25f;
  out[i]         = dE;                         // drugE
  out[S + i]     = sE;                         // disE
  out[2 * S + i] = 0.5f * de[i] + 0.5f * dE;   // drugAll (rd_drug == drug_emb)
  out[3 * S + i] = 0.5f * se[i] + 0.5f * sE;   // disAll
  out[4 * S + i] = de[i];                      // rd_drug
  out[5 * S + i] = se[i];                      // rd_dis
  if (i == 0) out[6 * S] = 0.f;                // meta_reg_loss
}

// ================================================================= launch
extern "C" void kernel_launch(void* const* d_in, const int* in_sizes, int n_in,
                              void* d_out, int out_size, void* d_ws, size_t ws_size,
                              hipStream_t stream) {
  const float* drug_emb = (const float*)d_in[0];
  const float* dis_emb  = (const float*)d_in[1];
  const float* Wr       = (const float*)d_in[2];
  const float* br       = (const float*)d_in[3];
  const float* Wd       = (const float*)d_in[4];
  const float* bd       = (const float*)d_in[5];
  const int*   rr_row   = (const int*)d_in[6];
  const int*   rr_col   = (const int*)d_in[7];
  const float* rr_val   = (const float*)d_in[8];
  const int*   dd_row   = (const int*)d_in[9];
  const int*   dd_col   = (const int*)d_in[10];
  const float* dd_val   = (const float*)d_in[11];
  const int*   rd_row   = (const int*)d_in[12];
  const int*   rd_col   = (const int*)d_in[13];
  const float* rd_val   = (const float*)d_in[14];
  const int E_rr = in_sizes[6], E_dd = in_sizes[9], E_rd = in_sizes[12];

  float* out = (float*)d_out;
  float* ws  = (float*)d_ws;

  constexpr size_t NF  = (size_t)DRUG_N * HID;   // 6.4M floats
  constexpr size_t NF2 = (size_t)NRD_N * HID;    // 12.8M floats
  float* cur_drug = ws;
  float* cur_dis  = ws + NF;
  float* cur_rd   = ws + 2 * NF;
  float* d0       = ws + 2 * NF + NF2;
  float* s0       = ws + 3 * NF + NF2;
  float* r0       = ws + 4 * NF + NF2;
  float* sum_drug = ws + 4 * NF + 2 * NF2;
  float* sum_dis  = ws + 5 * NF + 2 * NF2;
  float* deg      = ws + 6 * NF + 2 * NF2;
  float* deg_rr_r = deg;
  float* deg_rr_c = deg + DRUG_N;
  float* deg_dd_r = deg + 2 * DRUG_N;
  float* deg_dd_c = deg + 2 * DRUG_N + DIS_N;
  float* deg_rd_r = deg + 2 * DRUG_N + 2 * DIS_N;
  float* deg_rd_c = deg + 2 * DRUG_N + 2 * DIS_N + NRD_N;
  float* rd_stack = out + 6 * NF + 1;

  // degrees (graph-invariant across layers): zero then histogram
  zero_kernel<<<2048, 256, 0, stream>>>(deg, (size_t)(2 * DRUG_N + 2 * DIS_N + 2 * NRD_N));
  degree_kernel<<<(E_rr + 255) / 256, 256, 0, stream>>>(rr_row, rr_col, rr_val, deg_rr_r, deg_rr_c, E_rr);
  degree_kernel<<<(E_dd + 255) / 256, 256, 0, stream>>>(dd_row, dd_col, dd_val, deg_dd_r, deg_dd_c, E_dd);
  degree_kernel<<<(E_rd + 255) / 256, 256, 0, stream>>>(rd_row, rd_col, rd_val, deg_rd_r, deg_rd_c, E_rd);

  // gated embeddings via WMMA (writes cur_* and initializes sum_* = rr0/dd0)
  const int gateBlocks = ((DRUG_N / 16) + 7) / 8;
  gate_wmma_kernel<<<gateBlocks, 256, 0, stream>>>(drug_emb, Wr, br, cur_drug, sum_drug, DRUG_N);
  gate_wmma_kernel<<<gateBlocks, 256, 0, stream>>>(dis_emb,  Wd, bd, cur_dis,  sum_dis,  DIS_N);

  // cur_rd = concat(embs); rd_stack[:,0,:] = same
  init_rd_kernel<<<(int)((NF2 + 255) / 256), 256, 0, stream>>>(drug_emb, dis_emb, cur_rd, rd_stack);

  for (int i = 0; i < LAYERS; i++) {
    zero_kernel<<<4096, 256, 0, stream>>>(d0, 2 * NF + NF2);  // d0,s0,r0 contiguous
    spmm_kernel<<<(E_rr + 7) / 8, 256, 0, stream>>>(cur_drug, rr_row, rr_col, rr_val, deg_rr_r, deg_rr_c, d0, E_rr);
    spmm_kernel<<<(E_dd + 7) / 8, 256, 0, stream>>>(cur_dis,  dd_row, dd_col, dd_val, deg_dd_r, deg_dd_c, s0, E_dd);
    spmm_kernel<<<(E_rd + 7) / 8, 256, 0, stream>>>(cur_rd,   rd_row, rd_col, rd_val, deg_rd_r, deg_rd_c, r0, E_rd);
    layer_post_kernel<<<(NRD_N + 7) / 8, 256, 0, stream>>>(i, d0, s0, r0, cur_drug, cur_dis, cur_rd,
                                                           sum_drug, sum_dis, rd_stack);
  }

  final_kernel<<<(int)((NF + 255) / 256), 256, 0, stream>>>(drug_emb, dis_emb, sum_drug, sum_dis, out);
}
